// CustomMultiHeadAttention_4097398800711
// MI455X (gfx1250) — matrix operand order repaired and checked by reading, not compiled, not run on hardware
//
#include <hip/hip_runtime.h>

typedef _Float16 v16h __attribute__((ext_vector_type(16)));
typedef _Float16 v8h  __attribute__((ext_vector_type(8)));
typedef _Float16 v4h  __attribute__((ext_vector_type(4)));
typedef float    v8f  __attribute__((ext_vector_type(8)));
typedef int      v4i  __attribute__((ext_vector_type(4)));

#define D_MODEL  1024
#define NHEAD    16
#define HEAD_DIM 64
#define SEQ      2048
#define BATCH    2
#define M_TOTAL  (BATCH * SEQ)   // 4096

#define GLOBAL_AS __attribute__((address_space(1)))
#define LDS_AS    __attribute__((address_space(3)))

// ---- async global->LDS staging (CDNA5 GLOBAL_LOAD_ASYNC_TO_LDS_B128) ------
#if __has_builtin(__builtin_amdgcn_global_load_async_to_lds_b128)
#define ASYNC_LDS 1
static __device__ __forceinline__ void cp16_async(_Float16* l, const _Float16* g) {
  __builtin_amdgcn_global_load_async_to_lds_b128(
      (GLOBAL_AS v4i*)g, (LDS_AS v4i*)l, 0, 0);
}
static __device__ __forceinline__ void async_join() {
#if __has_builtin(__builtin_amdgcn_s_wait_asynccnt)
  __builtin_amdgcn_s_wait_asynccnt(0);
#else
  asm volatile("s_wait_asynccnt 0x0" ::: "memory");
#endif
}
#define ASYNC_JOIN() async_join()
#else
#define ASYNC_LDS 0
#define ASYNC_JOIN()
#endif

// Build a 16-f16 fragment from two 8-f16 (16B) LDS chunks.
// hiOff in f16 units: 16 for A-fragments (k runs at kbase and kbase+16),
// 16 for B-fragments as well, so A and B share one k order.
static __device__ __forceinline__ v16h frag2(const _Float16* p, int hiOff) {
  union { v16h v; v8h h2[2]; } u;
  u.h2[0] = *(const v8h*)p;
  u.h2[1] = *(const v8h*)(p + hiOff);
  return u.v;
}

// ---------------------------------------------------------------------------
// Weight fp32 -> fp16 conversion (vectorized x4)
// ---------------------------------------------------------------------------
__global__ void cvt_f32_to_f16_kernel(const float* __restrict__ s,
                                      _Float16* __restrict__ d, int n4) {
  int i  = blockIdx.x * blockDim.x + threadIdx.x;
  int st = blockDim.x * gridDim.x;
  for (; i < n4; i += st) {
    float4 f = ((const float4*)s)[i];
    v4h o = { (_Float16)f.x, (_Float16)f.y, (_Float16)f.z, (_Float16)f.w };
    ((v4h*)d)[i] = o;
  }
}

// ---------------------------------------------------------------------------
// Tiled NT GEMM via WMMA, computed TRANSPOSED:  C^T[n][m] = sum_k W[n,k]*A[m,k]
// (A-frags from the W tile, B-frags from the A tile). Each lane's C-fragment
// then holds one token m (= lane%16) and 8 contiguous model-dims n -> packed
// vector epilogue stores.
//   OUT_MODE 0: f16 head-major [B,H,T,hd]   (packed b128 stores)
//   OUT_MODE 1: f32 row-major  [M,N]        (packed float4 stores)
//   OUT_MODE 2: f16 head-major transposed [B,H,hd,T]  (V^T, scalar stores)
// Block 256 = 8 waves; block tile 128x128; wave (wr,wc) owns 64(n) x 32(m).
// K-tiles double-buffered; f16 streams staged with async LDS copies.
// ---------------------------------------------------------------------------
template<bool A_IS_F32, int OUT_MODE>
__global__ __launch_bounds__(256)
void gemm_nt_wmma(const void* __restrict__ Ain,
                  const _Float16* __restrict__ Bw,
                  const float* __restrict__ bias,
                  void* __restrict__ Cout, float outScale) {
  constexpr int K  = D_MODEL;
  constexpr int TM = 128, TN = 128, TK = 32;
  constexpr int LS = TK + 8;            // padded row stride (80B, 16B multiple)
  constexpr int BUF = TM * LS;

  __shared__ __align__(16) _Float16 As[2 * BUF];   // token rows [m][k]
  __shared__ __align__(16) _Float16 Bs[2 * BUF];   // weight rows [n][k]

  const int mBase = blockIdx.y * TM;
  const int nBase = blockIdx.x * TN;
  const int tid  = threadIdx.x;
  const int lane = tid & 31;
  const int wave = tid >> 5;
  const int wr   = wave >> 2;           // 0..1 -> n offset wr*64
  const int wc   = wave & 3;            // 0..3 -> m offset wc*32
  const int half = lane >> 4;
  const int l16  = lane & 15;

  auto stage = [&](int k0, int buf) {
    _Float16* Ad = &As[buf * BUF];
    _Float16* Bd = &Bs[buf * BUF];
#pragma unroll
    for (int i = 0; i < 2; i++) {
      int c  = tid + i * 256;           // 512 chunks of 8 f16
      int r  = c >> 2;                  // 0..127
      int cf = (c & 3) * 8;
      if (A_IS_F32) {
        const float* A  = (const float*)Ain;
        const float* gp = &A[(size_t)(mBase + r) * K + k0 + cf];
        float4 f0 = *(const float4*)gp;
        float4 f1 = *(const float4*)(gp + 4);
        v8h hv = { (_Float16)f0.x, (_Float16)f0.y, (_Float16)f0.z, (_Float16)f0.w,
                   (_Float16)f1.x, (_Float16)f1.y, (_Float16)f1.z, (_Float16)f1.w };
        *(v8h*)&Ad[r * LS + cf] = hv;
      } else {
        const _Float16* A = (const _Float16*)Ain;
#if ASYNC_LDS
        cp16_async(&Ad[r * LS + cf], &A[(size_t)(mBase + r) * K + k0 + cf]);
#else
        *(v8h*)&Ad[r * LS + cf] = *(const v8h*)&A[(size_t)(mBase + r) * K + k0 + cf];
#endif
      }
#if ASYNC_LDS
      cp16_async(&Bd[r * LS + cf], &Bw[(size_t)(nBase + r) * K + k0 + cf]);
#else
      *(v8h*)&Bd[r * LS + cf] = *(const v8h*)&Bw[(size_t)(nBase + r) * K + k0 + cf];
#endif
    }
  };

  v8f acc[4][2];
#pragma unroll
  for (int i = 0; i < 4; i++)
#pragma unroll
    for (int j = 0; j < 2; j++)
      acc[i][j] = (v8f){0.f,0.f,0.f,0.f,0.f,0.f,0.f,0.f};

  stage(0, 0);
  ASYNC_JOIN();
  __syncthreads();

  for (int k0 = 0; k0 < K; k0 += TK) {
    const int cur = (k0 / TK) & 1;
    if (k0 + TK < K) stage(k0 + TK, cur ^ 1);

    const _Float16* Ac = &As[cur * BUF];
    const _Float16* Bc = &Bs[cur * BUF];

    v16h af[4];                          // A-frags from W rows (n dim)
#pragma unroll
    for (int i = 0; i < 4; i++)
      af[i] = frag2(&Bc[(wr * 64 + i * 16 + l16) * LS + half * 8], 16);
    v16h bf[2];                          // B-frags from token rows (m dim)
#pragma unroll
    for (int j = 0; j < 2; j++)
      bf[j] = frag2(&Ac[(wc * 32 + j * 16 + l16) * LS + half * 8], 16);

#pragma unroll
    for (int i = 0; i < 4; i++)
#pragma unroll
      for (int j = 0; j < 2; j++)
        acc[i][j] = __builtin_amdgcn_wmma_f32_16x16x32_f16(
            false, af[i], false, bf[j], (short)0, acc[i][j], false, false);
    ASYNC_JOIN();
    __syncthreads();
  }

  // --- epilogue: C^T layout -> lane has token mCol, 8 contiguous n ----------
#pragma unroll
  for (int i = 0; i < 4; i++) {
    const int nRow0 = nBase + wr * 64 + i * 16 + half * 8;   // 8-aligned
    float4 b0 = *(const float4*)&bias[nRow0];
    float4 b1 = *(const float4*)&bias[nRow0 + 4];
    float bb[8] = { b0.x, b0.y, b0.z, b0.w, b1.x, b1.y, b1.z, b1.w };
#pragma unroll
    for (int j = 0; j < 2; j++) {
      const int mCol = mBase + wc * 32 + j * 16 + l16;
      if (OUT_MODE == 1) {
        float* op = &((float*)Cout)[(size_t)mCol * D_MODEL + nRow0];
        float4 o0 = { acc[i][j][0] + bb[0], acc[i][j][1] + bb[1],
                      acc[i][j][2] + bb[2], acc[i][j][3] + bb[3] };
        float4 o1 = { acc[i][j][4] + bb[4], acc[i][j][5] + bb[5],
                      acc[i][j][6] + bb[6], acc[i][j][7] + bb[7] };
        *(float4*)op = o0;
        *(float4*)(op + 4) = o1;
      } else if (OUT_MODE == 0) {
        int b = mCol >> 11, t = mCol & (SEQ - 1);
        int h = nRow0 >> 6, d0 = nRow0 & (HEAD_DIM - 1);   // run stays in head
        v8h hv;
#pragma unroll
        for (int r = 0; r < 8; r++)
          hv[r] = (_Float16)((acc[i][j][r] + bb[r]) * outScale);
        *(v8h*)&((_Float16*)Cout)[((size_t)((b * NHEAD + h) * SEQ + t)) * HEAD_DIM + d0] = hv;
      } else {                            // V^T [B,H,hd,T]
        int b = mCol >> 11, t = mCol & (SEQ - 1);
#pragma unroll
        for (int r = 0; r < 8; r++) {
          int n = nRow0 + r, h = n >> 6, d = n & (HEAD_DIM - 1);
          ((_Float16*)Cout)[((size_t)((b * NHEAD + h) * HEAD_DIM + d)) * SEQ + t] =
              (_Float16)((acc[i][j][r] + bb[r]) * outScale);
        }
      }
    }
  }
}

// ---------------------------------------------------------------------------
// Flash attention, fully transposed formulation.
//   S^T = K Q^T  (A-frags from K rows, B-frags from Q rows)
//   O^T = V^T P^T (A-frags from V^T rows, B-frags from P rows)
// Each lane's C-fragments hold ONE query (n = lane%16) -> softmax stats are
// scalar per lane with a single xor-16 shuffle; P writeback and context
// stores are packed 16B ops. Q pre-scaled by 1/8 in the Q projection.
// ---------------------------------------------------------------------------
__global__ __launch_bounds__(256)
void flash_attn_wmma(const _Float16* __restrict__ Q,   // [B*H, T, hd]
                     const _Float16* __restrict__ Kb,  // [B*H, T, hd]
                     const _Float16* __restrict__ Vb,  // [B*H, hd, T]
                     _Float16* __restrict__ ctx) {     // [B, T, D] f16
  constexpr int QT = 128, KT = 64;
  constexpr int LS  = HEAD_DIM + 8;                    // 144B stride
  constexpr int KBUF = KT * LS;
  constexpr int NKT  = SEQ / KT;

  __shared__ __align__(16) _Float16 Qs[QT * LS];       // reused as P scratch
  __shared__ __align__(16) _Float16 Ks[2 * KBUF];      // [key][d]
  __shared__ __align__(16) _Float16 Vt[2 * KBUF];      // [d][key]

  const int bh = blockIdx.y;
  const int qt = blockIdx.x;
  const int b  = bh / NHEAD, h = bh % NHEAD;
  const int tid = threadIdx.x, lane = tid & 31, wave = tid >> 5;
  const int half = lane >> 4, l16 = lane & 15;

  const size_t headBase = (size_t)bh * SEQ * HEAD_DIM;
  const _Float16* Qh = Q + headBase + (size_t)qt * QT * HEAD_DIM;

  auto stageKV = [&](int kti) {
    const int bo = (kti & 1) * KBUF;
    const _Float16* Kt = Kb + headBase + (size_t)kti * KT * HEAD_DIM;
    const _Float16* Vp = Vb + headBase + (size_t)kti * KT;   // column offset
#pragma unroll
    for (int i = 0; i < 2; i++) {
      int c = tid + i * 256;
      int r = c >> 3, cf = (c & 7) * 8;
#if ASYNC_LDS
      cp16_async(&Ks[bo + r * LS + cf], &Kt[(size_t)r * HEAD_DIM + cf]);
      cp16_async(&Vt[bo + r * LS + cf], &Vp[(size_t)r * SEQ + cf]);
#else
      *(v8h*)&Ks[bo + r * LS + cf] = *(const v8h*)&Kt[(size_t)r * HEAD_DIM + cf];
      *(v8h*)&Vt[bo + r * LS + cf] = *(const v8h*)&Vp[(size_t)r * SEQ + cf];
#endif
    }
  };

  // --- stage Q tile (128 x 64) and first K/V tile ---------------------------
#pragma unroll
  for (int i = 0; i < 4; i++) {
    int c = tid + i * 256;
    int r = c >> 3, cf = (c & 7) * 8;
#if ASYNC_LDS
    cp16_async(&Qs[r * LS + cf], &Qh[(size_t)r * HEAD_DIM + cf]);
#else
    *(v8h*)&Qs[r * LS + cf] = *(const v8h*)&Qh[(size_t)r * HEAD_DIM + cf];
#endif
  }
  stageKV(0);
  ASYNC_JOIN();
  __syncthreads();

  // Q as B-fragments: lane n = query l16, d = c*32 + 8*half + {0..7, 16..23}
  v16h qf[2];
#pragma unroll
  for (int c = 0; c < 2; c++)
    qf[c] = frag2(&Qs[(wave * 16 + l16) * LS + c * 32 + half * 8], 16);

  // wave's strip of Qs is now dead for this wave -> wave-private P scratch
  _Float16* Pw = &Qs[wave * 16 * LS];

  float mrow = -1e30f, lrow = 0.f;
  v8f o[4];
#pragma unroll
  for (int mt = 0; mt < 4; mt++) o[mt] = (v8f){0.f,0.f,0.f,0.f,0.f,0.f,0.f,0.f};

  for (int kt = 0; kt < NKT; kt++) {
    const int cur = kt & 1;
    if (kt + 1 < NKT) stageKV(kt + 1);            // overlap with compute
    const _Float16* Kc = &Ks[cur * KBUF];
    const _Float16* Vc = &Vt[cur * KBUF];

    // --- S^T = K Q^T : lane holds query l16, keys mt*16 + half*8 + r -------
    v8f s[4];
#pragma unroll
    for (int mt = 0; mt < 4; mt++) {
      v16h kf0 = frag2(&Kc[(mt * 16 + l16) * LS + 0 * 32 + half * 8], 16);
      v16h kf1 = frag2(&Kc[(mt * 16 + l16) * LS + 1 * 32 + half * 8], 16);
      v8f z = (v8f){0.f,0.f,0.f,0.f,0.f,0.f,0.f,0.f};
      z = __builtin_amdgcn_wmma_f32_16x16x32_f16(false, kf0, false, qf[0], (short)0, z, false, false);
      z = __builtin_amdgcn_wmma_f32_16x16x32_f16(false, kf1, false, qf[1], (short)0, z, false, false);
      s[mt] = z;
    }

    // --- online softmax: local 32-value reduce + one xor-16 shuffle --------
    float mloc = -1e30f;
#pragma unroll
    for (int mt = 0; mt < 4; mt++)
#pragma unroll
      for (int r = 0; r < 8; r++) mloc = fmaxf(mloc, s[mt][r]);
    mloc = fmaxf(mloc, __shfl_xor(mloc, 16, 32));
    float mnew = fmaxf(mrow, mloc);
    float fsc  = __expf(mrow - mnew);
    mrow = mnew;
    float ps = 0.f;
#pragma unroll
    for (int mt = 0; mt < 4; mt++)
#pragma unroll
      for (int r = 0; r < 8; r++) {
        float e = __expf(s[mt][r] - mnew);
        s[mt][r] = e;
        ps += e;
      }
    ps += __shfl_xor(ps, 16, 32);
    lrow = lrow * fsc + ps;

    // --- P writeback: packed b128 per m-tile (row q=l16, 8 contiguous keys)
#pragma unroll
    for (int mt = 0; mt < 4; mt++) {
      v8h ph;
#pragma unroll
      for (int r = 0; r < 8; r++) ph[r] = (_Float16)s[mt][r];
      *(v8h*)&Pw[l16 * LS + mt * 16 + half * 8] = ph;
    }
    asm volatile("s_wait_dscnt 0x0" ::: "memory");  // same-wave LDS RAW

    // P as B-fragments: lane n = query l16, keys c*32 + 8*half + {0..7, 16..23}
    v16h pf[2];
#pragma unroll
    for (int c = 0; c < 2; c++)
      pf[c] = frag2(&Pw[l16 * LS + c * 32 + half * 8], 16);

    // --- O^T = O^T*fsc + V^T P^T -------------------------------------------
#pragma unroll
    for (int mt = 0; mt < 4; mt++) {
#pragma unroll
      for (int r = 0; r < 8; r++) o[mt][r] *= fsc;
      v16h vf0 = frag2(&Vc[(mt * 16 + l16) * LS + 0 * 32 + half * 8], 16);
      v16h vf1 = frag2(&Vc[(mt * 16 + l16) * LS + 1 * 32 + half * 8], 16);
      o[mt] = __builtin_amdgcn_wmma_f32_16x16x32_f16(false, vf0, false, pf[0], (short)0, o[mt], false, false);
      o[mt] = __builtin_amdgcn_wmma_f32_16x16x32_f16(false, vf1, false, pf[1], (short)0, o[mt], false, false);
    }
    ASYNC_JOIN();
    __syncthreads();
  }

  // --- epilogue: lane owns one query row; packed b128 context stores --------
  const float recip = 1.0f / lrow;
  const int qrow = qt * QT + wave * 16 + l16;
  _Float16* cp = &ctx[((size_t)(b * SEQ + qrow)) * D_MODEL + h * HEAD_DIM];
#pragma unroll
  for (int mt = 0; mt < 4; mt++) {
    v8h oh;
#pragma unroll
    for (int r = 0; r < 8; r++) oh[r] = (_Float16)(o[mt][r] * recip);
    *(v8h*)&cp[mt * 16 + half * 8] = oh;
  }
}

// ---------------------------------------------------------------------------
extern "C" void kernel_launch(void* const* d_in, const int* in_sizes, int n_in,
                              void* d_out, int out_size, void* d_ws, size_t ws_size,
                              hipStream_t stream) {
  (void)in_sizes; (void)n_in; (void)out_size; (void)ws_size;
  const float* query = (const float*)d_in[0];
  const float* key   = (const float*)d_in[1];
  const float* value = (const float*)d_in[2];
  const float* Wq = (const float*)d_in[3];
  const float* bq = (const float*)d_in[4];
  const float* Wk = (const float*)d_in[5];
  const float* bk = (const float*)d_in[6];
  const float* Wv = (const float*)d_in[7];
  const float* bv = (const float*)d_in[8];
  const float* Wo = (const float*)d_in[9];
  const float* bo = (const float*)d_in[10];
  float* out = (float*)d_out;

  // workspace layout (all 16B-aligned, 40 MB total)
  const size_t WSZ = (size_t)D_MODEL * D_MODEL;   // 1M elems
  const size_t XSZ = (size_t)M_TOTAL * D_MODEL;   // 4M elems
  char* ws = (char*)d_ws;
  _Float16* Wq16  = (_Float16*)(ws);
  _Float16* Wk16  = Wq16 + WSZ;
  _Float16* Wv16  = Wk16 + WSZ;
  _Float16* Wo16  = Wv16 + WSZ;
  _Float16* Q16   = Wo16 + WSZ;     // [B,H,T,hd], pre-scaled by 1/8
  _Float16* K16   = Q16  + XSZ;     // [B,H,T,hd]
  _Float16* V16   = K16  + XSZ;     // [B,H,hd,T]  (pre-transposed)
  _Float16* ctx16 = V16  + XSZ;     // [B,T,D]

  const int n4 = (int)(WSZ / 4);
  cvt_f32_to_f16_kernel<<<512, 256, 0, stream>>>(Wq, Wq16, n4);
  cvt_f32_to_f16_kernel<<<512, 256, 0, stream>>>(Wk, Wk16, n4);
  cvt_f32_to_f16_kernel<<<512, 256, 0, stream>>>(Wv, Wv16, n4);
  cvt_f32_to_f16_kernel<<<512, 256, 0, stream>>>(Wo, Wo16, n4);

  dim3 gg(D_MODEL / 128, M_TOTAL / 128);   // (8, 32)
  gemm_nt_wmma<true, 0><<<gg, 256, 0, stream>>>(query, Wq16, bq, Q16, 0.125f);
  gemm_nt_wmma<true, 0><<<gg, 256, 0, stream>>>(key,   Wk16, bk, K16, 1.0f);
  gemm_nt_wmma<true, 2><<<gg, 256, 0, stream>>>(value, Wv16, bv, V16, 1.0f);

  dim3 ga(SEQ / 128, BATCH * NHEAD);       // (16, 32)
  flash_attn_wmma<<<ga, 256, 0, stream>>>(Q16, K16, V16, ctx16);

  gemm_nt_wmma<false, 1><<<gg, 256, 0, stream>>>(ctx16, Wo16, bo, out, 1.0f);
}
